// SelfAttention_57208964382852
// MI455X (gfx1250) — matrix operand order repaired and checked by reading, not compiled, not run on hardware
//
#include <hip/hip_runtime.h>
#include <hip/hip_bf16.h>

typedef __attribute__((ext_vector_type(16))) _Float16 v16h;
typedef __attribute__((ext_vector_type(8)))  _Float16 v8h;
typedef __attribute__((ext_vector_type(8)))  float    v8f;

#define EMBED 1024
#define HEADS 16
#define HD    64
#define BATCH 4
#define SEQ   2048
#define KC    64          // K/V columns processed per iteration
#define LPAD  80          // LDS row stride in halves (160B, keeps 32B alignment)

// workspace layout (units: _Float16 elements)
#define WS_QH   0u           // [N,H,S,hd] f16 : 8388608
#define WS_KH   8388608u     // [N,H,S,hd] f16
#define WS_VT   16777216u    // [N,H,hd,S] f16 (transposed V)
#define WS_OH   25165824u    // [N,S,E]    f16 (attention output)
#define WS_WOH  33554432u    // [E,E]      f16
#define WS_WQH  34603008u    // [hd,hd]
#define WS_WKH  34607104u
#define WS_WVH  34611200u

__device__ __forceinline__ v8f wmma_f16(v16h a, v16h b, v8f c) {
  return __builtin_amdgcn_wmma_f32_16x16x32_f16(false, a, false, b, (short)0, c,
                                                false, false);
}

__device__ __forceinline__ v16h cat8(v8h lo, v8h hi) {
  return __builtin_shufflevector(lo, hi, 0, 1, 2, 3, 4, 5, 6, 7,
                                 8, 9, 10, 11, 12, 13, 14, 15);
}

// Async DMA: 16 bytes/lane global -> LDS, tracked by ASYNCcnt (no VGPR data).
// Low 32 bits of a generic pointer to __shared__ are the LDS byte offset.
__device__ __forceinline__ void async_ld_b128(const _Float16* lds,
                                              const _Float16* g) {
  unsigned int       lo = (unsigned int)(unsigned long long)lds;
  unsigned long long ga = (unsigned long long)g;
  asm volatile("global_load_async_to_lds_b128 %0, %1, off"
               :: "v"(lo), "v"(ga) : "memory");
}

// ---------------------------------------------------------------------------
// Kernel 0: convert weights f32 -> f16 into workspace
// ---------------------------------------------------------------------------
__global__ void wconv_kernel(const float* __restrict__ Wv,
                             const float* __restrict__ Wk,
                             const float* __restrict__ Wq,
                             const float* __restrict__ Wo,
                             _Float16* __restrict__ ws) {
  int i = blockIdx.x * blockDim.x + threadIdx.x;
  if (i < EMBED * EMBED) ws[WS_WOH + i] = (_Float16)Wo[i];
  if (i < HD * HD) {
    ws[WS_WQH + i] = (_Float16)Wq[i];
    ws[WS_WKH + i] = (_Float16)Wk[i];
    ws[WS_WVH + i] = (_Float16)Wv[i];
  }
}

// ---------------------------------------------------------------------------
// Kernel 1: per-head QKV projection (y = x @ W^T per head, W is [hd,hd]).
// Q,K written [n,h,s,d]; V written transposed [n,h,d,s].
// ---------------------------------------------------------------------------
__device__ __forceinline__ void proj_one(const float* __restrict__ X,
                                         const _Float16* __restrict__ Wh,
                                         _Float16* __restrict__ dst,
                                         bool transposed, int n, int h, int l0,
                                         int lane) {
  const int m    = lane & 15;
  const int kb   = (lane >= 16) ? 8 : 0;
  const int ncol = lane & 15;
  const int db   = (lane >= 16) ? 16 : 0;
  const int ro   = (lane >= 16) ? 8 : 0;

  const float* p = X + ((size_t)(n * SEQ + l0 + m)) * EMBED + h * HD;
  v16h a0, a1;
#pragma unroll
  for (int j = 0; j < 8; ++j) {
    a0[j]     = (_Float16)p[db + j];
    a0[8 + j] = (_Float16)p[db + 8 + j];
    a1[j]     = (_Float16)p[32 + db + j];
    a1[8 + j] = (_Float16)p[40 + db + j];
  }
#pragma unroll
  for (int t = 0; t < 4; ++t) {
    const int e0 = t * 16;
    v16h b0 = *(const v16h*)(Wh + (e0 + ncol) * HD + db);
    v16h b1 = *(const v16h*)(Wh + (e0 + ncol) * HD + 32 + db);
    v8f c = {};
    c = wmma_f16(a0, b0, c);
    c = wmma_f16(a1, b1, c);
    if (!transposed) {
      _Float16* q = dst + ((size_t)(n * HEADS + h) * SEQ + l0) * HD;
#pragma unroll
      for (int r = 0; r < 8; ++r)
        q[(r + ro) * HD + e0 + ncol] = (_Float16)c[r];
    } else {
      _Float16* q = dst + (size_t)(n * HEADS + h) * HD * SEQ;
#pragma unroll
      for (int r = 0; r < 8; ++r)
        q[(size_t)(e0 + ncol) * SEQ + l0 + r + ro] = (_Float16)c[r];
    }
  }
}

__global__ __launch_bounds__(128) void qkv_proj_kernel(
    const float* __restrict__ Xq, const float* __restrict__ Xk,
    const float* __restrict__ Xv, _Float16* __restrict__ ws) {
  const int lane  = threadIdx.x & 31;
  const int wv    = threadIdx.x >> 5;
  const int b     = blockIdx.x;
  const int chunk = b & 31;
  const int h     = (b >> 5) & 15;
  const int n     = b >> 9;
  const int l0    = chunk * 64 + wv * 16;

  proj_one(Xq, ws + WS_WQH, ws + WS_QH, false, n, h, l0, lane);
  proj_one(Xk, ws + WS_WKH, ws + WS_KH, false, n, h, l0, lane);
  proj_one(Xv, ws + WS_WVH, ws + WS_VT, true,  n, h, l0, lane);
}

// ---------------------------------------------------------------------------
// Kernel 2: flash attention per (n,h). Block = 4 waves = 64 Q rows.
// K/V tiles (64x64) double-buffered in LDS via async-to-LDS DMA, shared by
// all 4 waves. Online softmax in base-2. 16 WMMAs / wave / iteration.
// ---------------------------------------------------------------------------
__device__ __forceinline__ void stage_kv(const _Float16* __restrict__ Kh,
                                         const _Float16* __restrict__ Vt,
                                         int k0, _Float16* kb_, _Float16* vb_,
                                         int tid) {
  // K tile: 64 rows x 128B, V tile: 64 rows x 128B; 16B per lane per op.
#pragma unroll
  for (int i = 0; i < 4; ++i) {
    int e   = i * 128 + tid;  // 0..511
    int row = e >> 3;
    int seg = e & 7;
    async_ld_b128(kb_ + row * LPAD + seg * 8,
                  Kh + (size_t)(k0 + row) * HD + seg * 8);
  }
#pragma unroll
  for (int i = 0; i < 4; ++i) {
    int e   = i * 128 + tid;
    int row = e >> 3;
    int seg = e & 7;
    async_ld_b128(vb_ + row * LPAD + seg * 8,
                  Vt + (size_t)row * SEQ + k0 + seg * 8);
  }
}

__global__ __launch_bounds__(128) void attn_kernel(_Float16* __restrict__ ws) {
  __shared__ _Float16 Kbuf[2][64 * LPAD];
  __shared__ _Float16 Vbuf[2][64 * LPAD];
  __shared__ _Float16 pbuf[4][16 * LPAD];

  const int tid   = threadIdx.x;
  const int lane  = tid & 31;
  const int wv    = tid >> 5;
  const int b     = blockIdx.x;
  const int chunk = b & 31;
  const int h     = (b >> 5) & 15;
  const int n     = b >> 9;
  const int q0    = chunk * 64 + wv * 16;

  const int m    = lane & 15;
  const int kb   = (lane >= 16) ? 8 : 0;
  const int ncol = lane & 15;
  const int db   = (lane >= 16) ? 16 : 0;
  const int ro   = (lane >= 16) ? 8 : 0;

  const _Float16* Qh = ws + WS_QH + (size_t)(n * HEADS + h) * SEQ * HD;
  const _Float16* Kh = ws + WS_KH + (size_t)(n * HEADS + h) * SEQ * HD;
  const _Float16* Vt = ws + WS_VT + (size_t)(n * HEADS + h) * HD * SEQ;

  v16h qa0, qa1;
  {
    const _Float16* p = Qh + (q0 + m) * HD;
    qa0 = cat8(*(const v8h*)(p + db), *(const v8h*)(p + 8 + db));
    qa1 = cat8(*(const v8h*)(p + 32 + db), *(const v8h*)(p + 40 + db));
  }

  v8f o[4] = {{}, {}, {}, {}};
  float mrow[8], lrow[8];
#pragma unroll
  for (int r = 0; r < 8; ++r) { mrow[r] = -3.0e38f; lrow[r] = 0.0f; }

  // (1/sqrt(EMBED)) * log2(e): do softmax in base 2 (v_exp_f32 is exp2)
  const float scale2 = 0.03125f * 1.44269504088896341f;

  stage_kv(Kh, Vt, 0, Kbuf[0], Vbuf[0], tid);

  const int NIT = SEQ / KC;
  for (int kc = 0; kc < NIT; ++kc) {
    const int cur = kc & 1;
    if (kc + 1 < NIT) {
      stage_kv(Kh, Vt, (kc + 1) * KC, Kbuf[1 - cur], Vbuf[1 - cur], tid);
      // leave only the 8 just-issued ops outstanding -> previous tile landed
      asm volatile("s_wait_asynccnt 0x8" ::: "memory");
    } else {
      asm volatile("s_wait_asynccnt 0x0" ::: "memory");
    }
    __syncthreads();

    // scores: 4 column tiles x (K=64 as 2 WMMAs)
    v8f s[4];
#pragma unroll
    for (int j = 0; j < 4; ++j) {
      const _Float16* kp = &Kbuf[cur][(j * 16 + ncol) * LPAD + db];
      v8f t = {};
      t = wmma_f16(qa0, *(const v16h*)(kp), t);
      t = wmma_f16(qa1, *(const v16h*)(kp + 32), t);
      s[j] = t;
    }

    float alpha[8];
#pragma unroll
    for (int r = 0; r < 8; ++r) {
      float x0 = s[0][r] * scale2, x1 = s[1][r] * scale2;
      float x2 = s[2][r] * scale2, x3 = s[3][r] * scale2;
      float rm = fmaxf(fmaxf(x0, x1), fmaxf(x2, x3));
      rm = fmaxf(rm, __shfl_xor(rm, 1, 32));
      rm = fmaxf(rm, __shfl_xor(rm, 2, 32));
      rm = fmaxf(rm, __shfl_xor(rm, 4, 32));
      rm = fmaxf(rm, __shfl_xor(rm, 8, 32));
      float mnew = fmaxf(mrow[r], rm);
      alpha[r] = __builtin_amdgcn_exp2f(mrow[r] - mnew);
      float p0 = __builtin_amdgcn_exp2f(x0 - mnew);
      float p1 = __builtin_amdgcn_exp2f(x1 - mnew);
      float p2 = __builtin_amdgcn_exp2f(x2 - mnew);
      float p3 = __builtin_amdgcn_exp2f(x3 - mnew);
      float rs = (p0 + p1) + (p2 + p3);
      rs += __shfl_xor(rs, 1, 32);
      rs += __shfl_xor(rs, 2, 32);
      rs += __shfl_xor(rs, 4, 32);
      rs += __shfl_xor(rs, 8, 32);
      lrow[r] = lrow[r] * alpha[r] + rs;
      mrow[r] = mnew;
      const int M = r + ro;
      pbuf[wv][M * LPAD + 0  + ncol] = (_Float16)p0;
      pbuf[wv][M * LPAD + 16 + ncol] = (_Float16)p1;
      pbuf[wv][M * LPAD + 32 + ncol] = (_Float16)p2;
      pbuf[wv][M * LPAD + 48 + ncol] = (_Float16)p3;
    }
#pragma unroll
    for (int r = 0; r < 8; ++r) {
      o[0][r] *= alpha[r]; o[1][r] *= alpha[r];
      o[2][r] *= alpha[r]; o[3][r] *= alpha[r];
    }

    // per-wave LDS ordering: P stores land before A-layout reads
    asm volatile("s_wait_dscnt 0x0" ::: "memory");

    v16h pa0, pa1;
    {
      const _Float16* pp = &pbuf[wv][m * LPAD];
      pa0 = cat8(*(const v8h*)(pp + db), *(const v8h*)(pp + 8 + db));
      pa1 = cat8(*(const v8h*)(pp + 32 + db), *(const v8h*)(pp + 40 + db));
    }

    // P(16x64) x V(64x64): 2 K-steps x 4 e-tiles
#pragma unroll
    for (int t = 0; t < 4; ++t) {
      const _Float16* vb = &Vbuf[cur][(t * 16 + ncol) * LPAD + db];
      o[t] = wmma_f16(pa0, *(const v16h*)(vb), o[t]);
      o[t] = wmma_f16(pa1, *(const v16h*)(vb + 32), o[t]);
    }

    // my ds reads of this buffer are done before anyone overwrites it
    asm volatile("s_wait_dscnt 0x0" ::: "memory");
    __syncthreads();
  }

  _Float16* Oh = ws + WS_OH;
#pragma unroll
  for (int r = 0; r < 8; ++r) {
    const float inv = 1.0f / lrow[r];
    const int M = r + ro;
    const size_t row = (size_t)(n * SEQ + q0 + M) * EMBED + h * HD;
    Oh[row + 0  + ncol] = (_Float16)(o[0][r] * inv);
    Oh[row + 16 + ncol] = (_Float16)(o[1][r] * inv);
    Oh[row + 32 + ncol] = (_Float16)(o[2][r] * inv);
    Oh[row + 48 + ncol] = (_Float16)(o[3][r] * inv);
  }
}

// ---------------------------------------------------------------------------
// Kernel 3: out = Oh @ Wo^T + bo, [8192,1024]x[1024,1024].
// Wo tile (64 cols x 64 K) double-buffered in LDS via async DMA, shared by
// all 4 waves. 128 WMMAs per wave.
// ---------------------------------------------------------------------------
__device__ __forceinline__ void stage_w(const _Float16* __restrict__ Woh,
                                        int eb, int c0, _Float16* wb_,
                                        int tid) {
#pragma unroll
  for (int i = 0; i < 4; ++i) {
    int e   = i * 128 + tid;
    int row = e >> 3;
    int seg = e & 7;
    async_ld_b128(wb_ + row * LPAD + seg * 8,
                  Woh + (size_t)(eb + row) * EMBED + c0 + seg * 8);
  }
}

__global__ __launch_bounds__(128) void oproj_kernel(
    const _Float16* __restrict__ ws, const float* __restrict__ bo,
    float* __restrict__ out) {
  __shared__ _Float16 Wbuf[2][64 * LPAD];

  const int tid  = threadIdx.x;
  const int lane = tid & 31;
  const int wv   = tid >> 5;
  const int t0   = blockIdx.x * 64 + wv * 16;
  const int eb   = blockIdx.y * 64;

  const int m    = lane & 15;
  const int kb   = (lane >= 16) ? 8 : 0;
  const int ncol = lane & 15;
  const int db   = (lane >= 16) ? 16 : 0;
  const int ro   = (lane >= 16) ? 8 : 0;

  const _Float16* Oh  = ws + WS_OH;
  const _Float16* Woh = ws + WS_WOH;

  v8f c[4] = {{}, {}, {}, {}};

  stage_w(Woh, eb, 0, Wbuf[0], tid);

  const int NIT = EMBED / KC;  // 16
  for (int it = 0; it < NIT; ++it) {
    const int cur = it & 1;
    const int c0  = it * KC;
    if (it + 1 < NIT) {
      stage_w(Woh, eb, c0 + KC, Wbuf[1 - cur], tid);
      asm volatile("s_wait_asynccnt 0x4" ::: "memory");
    } else {
      asm volatile("s_wait_asynccnt 0x0" ::: "memory");
    }
    __syncthreads();

    v16h a0, a1;
    {
      const _Float16* p = Oh + (size_t)(t0 + m) * EMBED + c0;
      a0 = cat8(*(const v8h*)(p + db), *(const v8h*)(p + 8 + db));
      a1 = cat8(*(const v8h*)(p + 32 + db), *(const v8h*)(p + 40 + db));
    }
#pragma unroll
    for (int t = 0; t < 4; ++t) {
      const _Float16* wp = &Wbuf[cur][(t * 16 + ncol) * LPAD + db];
      c[t] = wmma_f16(a0, *(const v16h*)(wp), c[t]);
      c[t] = wmma_f16(a1, *(const v16h*)(wp + 32), c[t]);
    }

    asm volatile("s_wait_dscnt 0x0" ::: "memory");
    __syncthreads();
  }

#pragma unroll
  for (int r = 0; r < 8; ++r) {
    const int M = r + ro;
    const size_t row = (size_t)(t0 + M) * EMBED + eb;
    out[row + 0  + ncol] = c[0][r] + bo[eb + 0  + ncol];
    out[row + 16 + ncol] = c[1][r] + bo[eb + 16 + ncol];
    out[row + 32 + ncol] = c[2][r] + bo[eb + 32 + ncol];
    out[row + 48 + ncol] = c[3][r] + bo[eb + 48 + ncol];
  }
}

// ---------------------------------------------------------------------------
extern "C" void kernel_launch(void* const* d_in, const int* in_sizes, int n_in,
                              void* d_out, int out_size, void* d_ws,
                              size_t ws_size, hipStream_t stream) {
  (void)in_sizes; (void)n_in; (void)out_size; (void)ws_size;
  const float* values = (const float*)d_in[0];
  const float* keys   = (const float*)d_in[1];
  const float* query  = (const float*)d_in[2];
  const float* Wv     = (const float*)d_in[3];
  const float* Wk     = (const float*)d_in[4];
  const float* Wq     = (const float*)d_in[5];
  const float* Wo     = (const float*)d_in[6];
  const float* bo     = (const float*)d_in[7];
  _Float16* ws = (_Float16*)d_ws;
  float* out   = (float*)d_out;

  wconv_kernel<<<4096, 256, 0, stream>>>(Wv, Wk, Wq, Wo, ws);
  qkv_proj_kernel<<<BATCH * HEADS * (SEQ / 64), 128, 0, stream>>>(query, keys,
                                                                  values, ws);
  attn_kernel<<<BATCH * HEADS * (SEQ / 64), 128, 0, stream>>>(ws);
  oproj_kernel<<<dim3((BATCH * SEQ) / 64, EMBED / 64), 128, 0, stream>>>(
      ws, bo, out);
}
